// MGE_28656021799513
// MI455X (gfx1250) — matrix operand order repaired and checked by reading, not compiled, not run on hardware
//
#include <hip/hip_runtime.h>

#define D     768
#define BB    2048
#define NDOM  9
#define NEXP  6
#define NSH   4
#define NTOT  10
#define BM    64                    // block row tile / domain padding granularity
#define BN    128                   // block col tile
#define BP_CAP (BB + NDOM*BM)       // 2624
#define ROW_TILES (BP_CAP/BM)       // 41
#define NCOL_BLK (D/BN)             // 6

typedef __bf16 bf16_t;
typedef __attribute__((ext_vector_type(16))) __bf16 v16bf;
typedef __attribute__((ext_vector_type(8)))  __bf16 v8bf;
typedef __attribute__((ext_vector_type(8)))  float  v8f;

// ---------------------------------------------------------------- utilities
__device__ __forceinline__ float silu_f(float x) { return x / (1.f + __expf(-x)); }

__device__ __forceinline__ v8f wmma_bf16(v16bf a, v16bf b, v8f c) {
  // D = A(16x32 bf16) * B(32x16 bf16) + C(16x16 f32)
  return __builtin_amdgcn_wmma_f32_16x16x32_bf16(false, a, false, b, (short)0, c,
                                                 false, false);
}

// A fragment: 16x32 tile, lane m=lane&15, h=lane>>4; K order {h*8..h*8+7, 16+h*8..+7}
__device__ __forceinline__ v16bf frag_a(const bf16_t (*As)[40], int m0, int lane) {
  int m = m0 + (lane & 15);
  int h = lane >> 4;
  v8bf lo = *(const v8bf*)&As[m][h * 8];
  v8bf hi = *(const v8bf*)&As[m][16 + h * 8];
  v16bf f;
#pragma unroll
  for (int i = 0; i < 8; i++) { f[i] = lo[i]; f[8 + i] = hi[i]; }
  return f;
}

// B fragment: 32x16 tile (stored transposed in LDS as [n][k]); lane n=lane&15,
// h=lane>>4; K order {h*8..h*8+7, 16+h*8..+7}
__device__ __forceinline__ v16bf frag_b(const bf16_t (*Bs)[40], int n0, int lane) {
  int n = n0 + (lane & 15);
  int h = lane >> 4;
  v8bf lo = *(const v8bf*)&Bs[n][h * 8];
  v8bf hi = *(const v8bf*)&Bs[n][16 + h * 8];
  v16bf f;
#pragma unroll
  for (int i = 0; i < 8; i++) { f[i] = lo[i]; f[8 + i] = hi[i]; }
  return f;
}

// Load 64(M) x 32(K) bf16 A tile into LDS (256 threads, one 16B chunk each)
__device__ __forceinline__ void load_A(const bf16_t* __restrict__ Ap, int lda,
                                       bf16_t (*As)[40], int tid) {
  int r = tid >> 2;          // 0..63
  int c = (tid & 3) * 8;     // 0,8,16,24 (bf16 elems)
  uint4 v = *(const uint4*)(Ap + (size_t)r * lda + c);
  *(uint4*)&As[r][c] = v;
}

// Load 32(K) x 128(N) f32 weight tile, convert->bf16, store transposed [n][k].
// One column per thread pair: coalesced b32 global loads, packed b128 LDS stores.
__device__ __forceinline__ void load_B(const float* __restrict__ Wp, int ldw, int col0,
                                       bf16_t (*Bs)[40], int tid, bool prefetch_next) {
  int n  = tid & 127;         // column within tile
  int kh = (tid >> 7) * 16;   // 0 or 16
  const float* src = Wp + (size_t)kh * ldw + col0 + n;
  v8bf t0, t1;
#pragma unroll
  for (int j = 0; j < 8; j++) t0[j] = (bf16_t)src[(size_t)j * ldw];
#pragma unroll
  for (int j = 0; j < 8; j++) t1[j] = (bf16_t)src[(size_t)(8 + j) * ldw];
  *(v8bf*)&Bs[n][kh]     = t0;
  *(v8bf*)&Bs[n][kh + 8] = t1;
  if (prefetch_next) __builtin_prefetch(src + (size_t)32 * ldw, 0, 0);
}

__device__ __forceinline__ int find_dom(const int* __restrict__ poff, int row0) {
  int n = 0;
  while (n < NDOM - 1 && row0 >= poff[n + 1]) n++;
  return n;
}

// ---------------------------------------------------------------- setup
__global__ void k_setup(const int* __restrict__ cat, int* poff, int* cursors) {
  if (threadIdx.x | blockIdx.x) return;
  int cnt[NDOM];
  for (int n = 0; n < NDOM; n++) cnt[n] = 0;
  for (int b = 0; b < BB; b++) { int c = cat[b]; if (c >= 0 && c < NDOM) cnt[c]++; }
  int acc = 0;
  poff[0] = 0;
  for (int n = 0; n < NDOM; n++) { acc += (cnt[n] + BM - 1) & ~(BM - 1); poff[n + 1] = acc; }
  for (int n = 0; n < NDOM; n++) cursors[n] = poff[n];
}

__global__ void k_perm_init(int* perm) {
  int p = blockIdx.x * blockDim.x + threadIdx.x;
  if (p < BP_CAP) perm[p] = -1;
}

__global__ void k_scatter(const int* __restrict__ cat, int* cursors, int* perm) {
  int b = blockIdx.x * blockDim.x + threadIdx.x;
  if (b >= BB) return;
  int n = cat[b];
  int pos = atomicAdd(&cursors[n], 1);
  perm[pos] = b;
}

// gather x into permuted bf16 Xp[BP_CAP][D] and gate input Gp[BP_CAP][2D]=[demb;x]
__global__ void k_gather(const float* __restrict__ x, const float* __restrict__ demb,
                         const int* __restrict__ cat, const int* __restrict__ perm,
                         bf16_t* __restrict__ Xp, bf16_t* __restrict__ Gp) {
  int p = blockIdx.x;
  int b = perm[p];
  for (int d = threadIdx.x; d < D; d += blockDim.x) {
    float xv = 0.f, ev = 0.f;
    if (b >= 0) {
      xv = x[(size_t)b * D + d];
      ev = demb[(size_t)cat[b] * D + d];
    }
    bf16_t xb = (bf16_t)xv;
    Xp[(size_t)p * D + d]            = xb;
    Gp[(size_t)p * 2 * D + d]        = (bf16_t)ev;
    Gp[(size_t)p * 2 * D + D + d]    = xb;
  }
}

// ---------------------------------------------------------------- gate hidden GEMM
// Hg = silu(Gp[rows_of_dom] * gW1[dom] + gb1[dom]) , K = 2D
__global__ void k_gemm_gateh(const bf16_t* __restrict__ Gp, const float* __restrict__ gW1,
                             const float* __restrict__ gb1, const int* __restrict__ poff,
                             bf16_t* __restrict__ Hg) {
  __shared__ __align__(16) bf16_t As[BM][40];
  __shared__ __align__(16) bf16_t Bs[BN][40];
  int row0 = blockIdx.y * BM;
  if (row0 >= poff[NDOM]) return;
  int dom  = find_dom(poff, row0);
  int col0 = blockIdx.x * BN;
  const float* W = gW1 + (size_t)dom * (2 * D) * D;
  int tid = threadIdx.x, wave = tid >> 5, lane = tid & 31;
  int wm = wave >> 2, wn = wave & 3;   // 2 x 4 wave grid
  v8f acc[2][2] = {};
  for (int k0 = 0; k0 < 2 * D; k0 += 32) {
    load_A(Gp + (size_t)row0 * (2 * D) + k0, 2 * D, As, tid);
    load_B(W + (size_t)k0 * D, D, col0, Bs, tid, k0 + 32 < 2 * D);
    __syncthreads();
    v16bf a0 = frag_a(As, wm * 32, lane), a1 = frag_a(As, wm * 32 + 16, lane);
    v16bf b0 = frag_b(Bs, wn * 32, lane), b1 = frag_b(Bs, wn * 32 + 16, lane);
    acc[0][0] = wmma_bf16(a0, b0, acc[0][0]);
    acc[0][1] = wmma_bf16(a0, b1, acc[0][1]);
    acc[1][0] = wmma_bf16(a1, b0, acc[1][0]);
    acc[1][1] = wmma_bf16(a1, b1, acc[1][1]);
    __syncthreads();
  }
  const float* bias = gb1 + (size_t)dom * D;
#pragma unroll
  for (int mt = 0; mt < 2; mt++)
#pragma unroll
    for (int nt = 0; nt < 2; nt++)
#pragma unroll
      for (int r = 0; r < 8; r++) {
        int grow = row0 + wm * 32 + mt * 16 + (lane >> 4) * 8 + r;
        int gcol = col0 + wn * 32 + nt * 16 + (lane & 15);
        Hg[(size_t)grow * D + gcol] = (bf16_t)silu_f(acc[mt][nt][r] + bias[gcol]);
      }
}

// ---------------------------------------------------------------- gate logits/softmax
__global__ void k_gate_logits(const bf16_t* __restrict__ Hg, const float* __restrict__ gW2,
                              const float* __restrict__ gb2, const float* __restrict__ temp,
                              const int* __restrict__ perm, const int* __restrict__ cat,
                              float* __restrict__ gates) {
  int p = blockIdx.x * 8 + (threadIdx.x >> 5);
  int lane = threadIdx.x & 31;
  if (p >= BP_CAP) return;
  int b = perm[p];
  if (b < 0) {
    if (lane < NTOT) gates[(size_t)p * NTOT + lane] = 0.f;
    return;
  }
  int n = cat[b];
  const float* W = gW2 + (size_t)n * D * NTOT;
  float acc[NTOT];
#pragma unroll
  for (int e = 0; e < NTOT; e++) acc[e] = 0.f;
  for (int k = lane; k < D; k += 32) {
    float h = (float)Hg[(size_t)p * D + k];
    const float* wr = W + (size_t)k * NTOT;
#pragma unroll
    for (int e = 0; e < NTOT; e++) acc[e] += h * wr[e];
  }
#pragma unroll
  for (int off = 16; off > 0; off >>= 1)
#pragma unroll
    for (int e = 0; e < NTOT; e++) acc[e] += __shfl_down(acc[e], off, 32);
  if (lane == 0) {
    float tn = logf(1.f + __expf(temp[n])) + 1e-4f;  // softplus + eps
    float mx = -1e30f;
#pragma unroll
    for (int e = 0; e < NTOT; e++) {
      acc[e] = (acc[e] + gb2[n * NTOT + e]) / tn;
      mx = fmaxf(mx, acc[e]);
    }
    float s = 0.f;
#pragma unroll
    for (int e = 0; e < NTOT; e++) { acc[e] = __expf(acc[e] - mx); s += acc[e]; }
    float inv = 1.f / s;
#pragma unroll
    for (int e = 0; e < NTOT; e++) gates[(size_t)p * NTOT + e] = acc[e] * inv;
  }
}

// ---------------------------------------------------------------- expert layer-1 GEMM
// H_t = gate_t * silu(Xp * W1_t + b1_t);  t = blockIdx.z in [0,10)
__global__ void k_gemm_l1(const bf16_t* __restrict__ Xp,
                          const float* __restrict__ W1dom, const float* __restrict__ b1dom,
                          const float* __restrict__ W1sh,  const float* __restrict__ b1sh,
                          const float* __restrict__ gates, const int* __restrict__ poff,
                          bf16_t* __restrict__ Hbuf) {
  __shared__ __align__(16) bf16_t As[BM][40];
  __shared__ __align__(16) bf16_t Bs[BN][40];
  int row0 = blockIdx.y * BM;
  if (row0 >= poff[NDOM]) return;
  int dom  = find_dom(poff, row0);
  int col0 = blockIdx.x * BN;
  int t    = blockIdx.z;
  const float* W    = (t < NEXP) ? W1dom + (size_t)(dom * NEXP + t) * D * D
                                 : W1sh  + (size_t)(t - NEXP) * D * D;
  const float* bias = (t < NEXP) ? b1dom + (size_t)(dom * NEXP + t) * D
                                 : b1sh  + (size_t)(t - NEXP) * D;
  int tid = threadIdx.x, wave = tid >> 5, lane = tid & 31;
  int wm = wave >> 2, wn = wave & 3;
  v8f acc[2][2] = {};
  for (int k0 = 0; k0 < D; k0 += 32) {
    load_A(Xp + (size_t)row0 * D + k0, D, As, tid);
    load_B(W + (size_t)k0 * D, D, col0, Bs, tid, k0 + 32 < D);
    __syncthreads();
    v16bf a0 = frag_a(As, wm * 32, lane), a1 = frag_a(As, wm * 32 + 16, lane);
    v16bf b0 = frag_b(Bs, wn * 32, lane), b1 = frag_b(Bs, wn * 32 + 16, lane);
    acc[0][0] = wmma_bf16(a0, b0, acc[0][0]);
    acc[0][1] = wmma_bf16(a0, b1, acc[0][1]);
    acc[1][0] = wmma_bf16(a1, b0, acc[1][0]);
    acc[1][1] = wmma_bf16(a1, b1, acc[1][1]);
    __syncthreads();
  }
  bf16_t* Ht = Hbuf + (size_t)t * BP_CAP * D;
#pragma unroll
  for (int mt = 0; mt < 2; mt++)
#pragma unroll
    for (int nt = 0; nt < 2; nt++)
#pragma unroll
      for (int r = 0; r < 8; r++) {
        int grow = row0 + wm * 32 + mt * 16 + (lane >> 4) * 8 + r;
        int gcol = col0 + wn * 32 + nt * 16 + (lane & 15);
        float g  = gates[(size_t)grow * NTOT + t];
        float v  = silu_f(acc[mt][nt][r] + bias[gcol]) * g;
        Ht[(size_t)grow * D + gcol] = (bf16_t)v;
      }
}

// ---------------------------------------------------------------- output GEMM
// out = sum_t H_t * W2_t + sum_t gate_t * b2_t ; scatter to original rows
__global__ void k_gemm_out(const bf16_t* __restrict__ Hbuf,
                           const float* __restrict__ W2dom, const float* __restrict__ b2dom,
                           const float* __restrict__ W2sh,  const float* __restrict__ b2sh,
                           const float* __restrict__ gates, const int* __restrict__ poff,
                           const int* __restrict__ perm, float* __restrict__ out,
                           int out_col0) {
  __shared__ __align__(16) bf16_t As[BM][40];
  __shared__ __align__(16) bf16_t Bs[BN][40];
  int row0 = blockIdx.y * BM;
  if (row0 >= poff[NDOM]) return;
  int dom  = find_dom(poff, row0);
  int col0 = blockIdx.x * BN;
  int tid = threadIdx.x, wave = tid >> 5, lane = tid & 31;
  int wm = wave >> 2, wn = wave & 3;
  v8f acc[2][2] = {};
  for (int t = 0; t < NTOT; t++) {
    const bf16_t* A = Hbuf + (size_t)t * BP_CAP * D + (size_t)row0 * D;
    const float*  W = (t < NEXP) ? W2dom + (size_t)(dom * NEXP + t) * D * D
                                 : W2sh  + (size_t)(t - NEXP) * D * D;
    for (int k0 = 0; k0 < D; k0 += 32) {
      load_A(A + k0, D, As, tid);
      load_B(W + (size_t)k0 * D, D, col0, Bs, tid, k0 + 32 < D);
      __syncthreads();
      v16bf a0 = frag_a(As, wm * 32, lane), a1 = frag_a(As, wm * 32 + 16, lane);
      v16bf b0 = frag_b(Bs, wn * 32, lane), b1 = frag_b(Bs, wn * 32 + 16, lane);
      acc[0][0] = wmma_bf16(a0, b0, acc[0][0]);
      acc[0][1] = wmma_bf16(a0, b1, acc[0][1]);
      acc[1][0] = wmma_bf16(a1, b0, acc[1][0]);
      acc[1][1] = wmma_bf16(a1, b1, acc[1][1]);
      __syncthreads();
    }
  }
#pragma unroll
  for (int mt = 0; mt < 2; mt++)
#pragma unroll
    for (int nt = 0; nt < 2; nt++)
#pragma unroll
      for (int r = 0; r < 8; r++) {
        int grow = row0 + wm * 32 + mt * 16 + (lane >> 4) * 8 + r;
        int gcol = col0 + wn * 32 + nt * 16 + (lane & 15);
        int b = perm[grow];
        if (b < 0) continue;
        const float* g = gates + (size_t)grow * NTOT;
        float bias = 0.f;
#pragma unroll
        for (int t = 0; t < NEXP; t++)
          bias += g[t] * b2dom[(size_t)(dom * NEXP + t) * D + gcol];
#pragma unroll
        for (int t = 0; t < NSH; t++)
          bias += g[NEXP + t] * b2sh[(size_t)t * D + gcol];
        out[(size_t)b * (2 * D) + out_col0 + gcol] = acc[mt][nt][r] + bias;
      }
}

// ---------------------------------------------------------------- launch
extern "C" void kernel_launch(void* const* d_in, const int* in_sizes, int n_in,
                              void* d_out, int out_size, void* d_ws, size_t ws_size,
                              hipStream_t stream) {
  const float* xin[2]  = {(const float*)d_in[0], (const float*)d_in[1]};
  const int*   cat     = (const int*)d_in[2];
  const float* demb    = (const float*)d_in[3];
  const float *W1[2], *b1[2], *W2[2], *b2[2], *sW1[2], *sb1[2], *sW2[2], *sb2[2];
  const float *gW1[2], *gb1[2], *gW2[2], *gb2[2], *temp[2];
  for (int br = 0; br < 2; br++) {
    int base = 4 + br * 13;
    W1[br]  = (const float*)d_in[base + 0];
    b1[br]  = (const float*)d_in[base + 1];
    W2[br]  = (const float*)d_in[base + 2];
    b2[br]  = (const float*)d_in[base + 3];
    sW1[br] = (const float*)d_in[base + 4];
    sb1[br] = (const float*)d_in[base + 5];
    sW2[br] = (const float*)d_in[base + 6];
    sb2[br] = (const float*)d_in[base + 7];
    gW1[br] = (const float*)d_in[base + 8];
    gb1[br] = (const float*)d_in[base + 9];
    gW2[br] = (const float*)d_in[base + 10];
    gb2[br] = (const float*)d_in[base + 11];
    temp[br]= (const float*)d_in[base + 12];
  }

  // workspace carve (~57 MB total)
  char* w = (char*)d_ws;
  int* poff    = (int*)w;        // 10 ints
  int* cursors = poff + 16;      // 9 ints
  int* perm    = cursors + 16;   // BP_CAP ints
  size_t o = (((size_t)(32 + BP_CAP) * 4) + 255) & ~(size_t)255;
  bf16_t* Xp = (bf16_t*)(w + o); o += (size_t)BP_CAP * D * 2;
  bf16_t* Gp = (bf16_t*)(w + o); o += (size_t)BP_CAP * 2 * D * 2;
  bf16_t* Hg = (bf16_t*)(w + o); o += (size_t)BP_CAP * D * 2;
  float* gates = (float*)(w + o); o += (((size_t)BP_CAP * NTOT * 4) + 255) & ~(size_t)255;
  bf16_t* Hbuf = (bf16_t*)(w + o); o += (size_t)NTOT * BP_CAP * D * 2;
  (void)o; (void)ws_size; (void)in_sizes; (void)n_in; (void)out_size;

  k_setup<<<1, 1, 0, stream>>>(cat, poff, cursors);
  k_perm_init<<<(BP_CAP + 255) / 256, 256, 0, stream>>>(perm);
  k_scatter<<<(BB + 255) / 256, 256, 0, stream>>>(cat, cursors, perm);

  float* outp = (float*)d_out;
  dim3 g2(NCOL_BLK, ROW_TILES);
  dim3 g3(NCOL_BLK, ROW_TILES, NTOT);
  for (int br = 0; br < 2; br++) {
    k_gather<<<BP_CAP, 256, 0, stream>>>(xin[br], demb, cat, perm, Xp, Gp);
    k_gemm_gateh<<<g2, 256, 0, stream>>>(Gp, gW1[br], gb1[br], poff, Hg);
    k_gate_logits<<<(BP_CAP + 7) / 8, 256, 0, stream>>>(Hg, gW2[br], gb2[br], temp[br],
                                                        perm, cat, gates);
    k_gemm_l1<<<g3, 256, 0, stream>>>(Xp, W1[br], b1[br], sW1[br], sb1[br], gates, poff,
                                      Hbuf);
    k_gemm_out<<<g2, 256, 0, stream>>>(Hbuf, W2[br], b2[br], sW2[br], sb2[br], gates,
                                       poff, perm, outp, br * D);
  }
}
